// Seq2SeqVAE_60129542144644
// MI455X (gfx1250) — matrix operand order repaired and checked by reading, not compiled, not run on hardware
//
#include <hip/hip_runtime.h>
#include <hip/hip_bf16.h>

// Seq2SeqVAE for MI455X (gfx1250): fused bf16-WMMA GRU step kernels.
// v2: 2 M-subtiles per wave (32-row blocks) -> each weight B-fragment feeds
// 2 WMMAs, halving L2 weight traffic of the bandwidth-bound recurrent steps.

typedef __attribute__((ext_vector_type(16))) __bf16 v16bf;
typedef __attribute__((ext_vector_type(8)))  __bf16 v8bf;
typedef __attribute__((ext_vector_type(8)))  float  v8f;

#define BB  256   // batch
#define TT  128   // time steps
#define DIN 128   // input dim
#define HHE 512   // encoder hidden per direction
#define HHD 256   // decoder hidden per direction
#define QN  (BB * 2048)   // q output elements (B x 2H)

__device__ __forceinline__ float fsig(float x) { return 1.0f / (1.0f + __expf(-x)); }
__device__ __forceinline__ float ftanh_fast(float x) {
  float e = __expf(2.0f * x);
  return (e - 1.0f) / (e + 1.0f);
}

// A-fragment (16x32 bf16, M x K) from a row-major panel with arbitrary row
// stride (elements). Works for both LDS panels and global activation rows.
// lanes 0-15 hold K {k0+0..15}, lanes 16-31 hold {k0+16..31}.
__device__ __forceinline__ v16bf afrag(const __bf16* base, long strideE, int sub,
                                       int k0, int lane) {
  const __bf16* p = base + (size_t)((lane & 15) + sub * 16) * (size_t)strideE
                        + (size_t)(k0 + ((lane >> 4) << 4));
  v16bf a;
#pragma unroll
  for (int i = 0; i < 8; ++i) { a[i] = p[i]; a[8 + i] = p[8 + i]; }
  return a;
}

// B-fragment (32x16 bf16, K x N) directly from row-major weight W[N,K]:
// lane n = lane&15, 16 contiguous K values at k0 + (lane>>4)*16 -> one 32B load.
__device__ __forceinline__ v16bf gBfrag(const __bf16* W, int K, int wrow, int k0, int lane) {
  const __bf16* p = W + (size_t)wrow * K + (size_t)(k0 + ((lane >> 4) << 4));
  return *(const v16bf*)p;
}

#define WMMA_BF16(a, b, c) \
  __builtin_amdgcn_wmma_f32_16x16x32_bf16(false, (a), false, (b), (short)0, (c), false, false)

__global__ void f32_to_bf16_k(const float* __restrict__ s, __bf16* __restrict__ d, int n) {
  int i = blockIdx.x * 256 + threadIdx.x;
  if (i < n) d[i] = (__bf16)s[i];
}

// One fused bidirectional GRU time step.
//   gates = X·Wih^T + bih , H·Whh^T + bhh ; r,z merged over both K ranges,
//   n-gate kept split (i_n vs h_n). grid = (B/32, Nh/128, 2 dirs), 256 thr.
__global__ __launch_bounds__(256) void gru_step_fused(
    const __bf16* __restrict__ Xin, long xs_t, long xs_b, int t, int Tlen,
    const __bf16* __restrict__ Wih, const __bf16* __restrict__ Whh,
    const float* __restrict__ bih, const float* __restrict__ bhh,
    const float* __restrict__ hcurF, const __bf16* __restrict__ hcurB,
    float* __restrict__ hnxtF, __bf16* __restrict__ hnxtB,
    __bf16* __restrict__ ys, long ys_t, long ys_b,
    int Kin, int Nh)
{
  __shared__ __bf16 sH[32 * 512];   // hidden panel, 32 rows, reused by 8 waves

  const int tid = threadIdx.x;
  const int dir = blockIdx.z;
  const int te  = dir ? (Tlen - 1 - t) : t;   // backward direction walks reversed
  const int m0  = blockIdx.x * 32;

  const __bf16* WihD = Wih + (size_t)dir * 3u * Nh * Kin;
  const __bf16* WhhD = Whh + (size_t)dir * 3u * Nh * Nh;
  const float*  bihD = bih + (size_t)dir * 3u * Nh;
  const float*  bhhD = bhh + (size_t)dir * 3u * Nh;
  const float*  hcF  = hcurF + (size_t)dir * BB * Nh;
  const __bf16* hcB  = hcurB + (size_t)dir * BB * Nh;
  float*        hnF  = hnxtF + (size_t)dir * BB * Nh;
  __bf16*       hnB  = hnxtB + (size_t)dir * BB * Nh;

  // Stage 32-row hidden panel into LDS (8-way reuse across waves).
  {
    const int vh = Nh >> 3;
    for (int r = 0; r < 32; ++r) {
      const v8bf* src = (const v8bf*)(hcB + (size_t)(m0 + r) * Nh);
      v8bf* dst = (v8bf*)(sH + r * Nh);
      for (int c = tid; c < vh; c += 256) dst[c] = src[c];
    }
  }
  __syncthreads();

  const int wave = tid >> 5;
  const int lane = tid & 31;
  const int n0w  = blockIdx.y * 128 + wave * 16;
  const int ncol = n0w + (lane & 15);

  // Input activation panel read directly from global (hits WGP$ across waves).
  const __bf16* Abase = Xin + (size_t)te * xs_t + (size_t)m0 * xs_b;

  v8f ar0 = {}, ar1 = {}, az0 = {}, az1 = {};
  v8f ai0 = {}, ai1 = {}, ah0 = {}, ah1 = {};

  // Input-projection K range: accumulates r, z, i_n for both M-subtiles.
  for (int k = 0; k < Kin; k += 32) {
    v16bf a0 = afrag(Abase, xs_b, 0, k, lane);
    v16bf a1 = afrag(Abase, xs_b, 1, k, lane);
    v16bf br = gBfrag(WihD, Kin, ncol,          k, lane);
    v16bf bz = gBfrag(WihD, Kin, Nh + ncol,     k, lane);
    v16bf bn = gBfrag(WihD, Kin, 2 * Nh + ncol, k, lane);
    __builtin_prefetch(WihD + (size_t)ncol * Kin + k + 256, 0, 1);
    ar0 = WMMA_BF16(a0, br, ar0);  ar1 = WMMA_BF16(a1, br, ar1);
    az0 = WMMA_BF16(a0, bz, az0);  az1 = WMMA_BF16(a1, bz, az1);
    ai0 = WMMA_BF16(a0, bn, ai0);  ai1 = WMMA_BF16(a1, bn, ai1);
  }
  // Recurrent K range: accumulates r, z, h_n for both M-subtiles.
  for (int k = 0; k < Nh; k += 32) {
    v16bf a0 = afrag(sH, Nh, 0, k, lane);
    v16bf a1 = afrag(sH, Nh, 1, k, lane);
    v16bf br = gBfrag(WhhD, Nh, ncol,          k, lane);
    v16bf bz = gBfrag(WhhD, Nh, Nh + ncol,     k, lane);
    v16bf bn = gBfrag(WhhD, Nh, 2 * Nh + ncol, k, lane);
    __builtin_prefetch(WhhD + (size_t)ncol * Nh + k + 256, 0, 1);
    ar0 = WMMA_BF16(a0, br, ar0);  ar1 = WMMA_BF16(a1, br, ar1);
    az0 = WMMA_BF16(a0, bz, az0);  az1 = WMMA_BF16(a1, bz, az1);
    ah0 = WMMA_BF16(a0, bn, ah0);  ah1 = WMMA_BF16(a1, bn, ah1);
  }

  // Gate epilogue: torch GRUCell math, f32 state update, bf16 copy for next step.
  const float b_r  = bihD[ncol] + bhhD[ncol];
  const float b_z  = bihD[Nh + ncol] + bhhD[Nh + ncol];
  const float bi_n = bihD[2 * Nh + ncol];
  const float bh_n = bhhD[2 * Nh + ncol];

#pragma unroll
  for (int s = 0; s < 2; ++s) {
    const v8f vr = s ? ar1 : ar0;
    const v8f vz = s ? az1 : az0;
    const v8f vi = s ? ai1 : ai0;
    const v8f vh = s ? ah1 : ah0;
    const int mbase = m0 + s * 16 + ((lane >> 4) << 3);
#pragma unroll
    for (int v = 0; v < 8; ++v) {
      const int brow = mbase + v;
      float r = fsig(vr[v] + b_r);
      float z = fsig(vz[v] + b_z);
      float n = ftanh_fast(vi[v] + bi_n + r * (vh[v] + bh_n));
      float hold = hcF[(size_t)brow * Nh + ncol];
      float hnew = (1.0f - z) * n + z * hold;
      hnF[(size_t)brow * Nh + ncol] = hnew;
      __bf16 hb = (__bf16)hnew;
      hnB[(size_t)brow * Nh + ncol] = hb;
      if (ys) ys[(size_t)te * ys_t + (size_t)brow * ys_b + (size_t)dir * Nh + ncol] = hb;
    }
  }
}

// Decoder output head: rec[:,t,:] = Y·linW^T + b ; also feeds back as next input.
__global__ __launch_bounds__(256) void linear_step(
    const __bf16* __restrict__ Y, const __bf16* __restrict__ W,
    const float* __restrict__ bias, float* __restrict__ rec,
    __bf16* __restrict__ inp, int t)
{
  __shared__ __bf16 sA[16 * 512];
  const int tid = threadIdx.x;
  const int m0  = blockIdx.x * 16;
  for (int r = 0; r < 16; ++r) {
    const v8bf* src = (const v8bf*)(Y + (size_t)(m0 + r) * 512);
    v8bf* dst = (v8bf*)(sA + r * 512);
    for (int c = tid; c < 64; c += 256) dst[c] = src[c];
  }
  __syncthreads();

  const int wave = tid >> 5;
  const int lane = tid & 31;
  const int ncol = wave * 16 + (lane & 15);   // 8 waves cover all 128 outputs

  v8f acc = {};
  for (int k = 0; k < 512; k += 32) {
    v16bf a = afrag(sA, 512, 0, k, lane);
    v16bf b = gBfrag(W, 512, ncol, k, lane);
    acc = WMMA_BF16(a, b, acc);
  }
  const float bb = bias[ncol];
  const int mbase = m0 + ((lane >> 4) << 3);
#pragma unroll
  for (int v = 0; v < 8; ++v) {
    const int brow = mbase + v;
    float val = acc[v] + bb;
    rec[((size_t)brow * TT + t) * DIN + ncol] = val;   // rec is [B,T,DIN]
    inp[(size_t)brow * DIN + ncol] = (__bf16)val;
  }
}

// q output + reparameterized latent -> decoder initial hidden; zero decoder input.
__global__ void latent_k(const float* __restrict__ hE0, const float* __restrict__ hE1,
                         const float* __restrict__ eps, float* __restrict__ q,
                         float* __restrict__ hidF, __bf16* __restrict__ hidB,
                         __bf16* __restrict__ inp)
{
  int i = blockIdx.x * 256 + threadIdx.x;   // 0 .. B*1024-1
  if (i >= BB * 1024) return;
  int b = i >> 10, c = i & 1023;
  int dd = c >> 9, j = c & 511;
  float mu = hE0[(size_t)dd * BB * HHE + (size_t)b * HHE + j];   // layer0 finals
  float lv = hE1[(size_t)dd * BB * HHE + (size_t)b * HHE + j];   // layer1 finals
  q[(size_t)b * 2048 + c]        = mu;
  q[(size_t)b * 2048 + 1024 + c] = lv;
  float lat = mu + __expf(lv) * eps[i];
  hidF[i] = lat;                 // flat reinterpret == torch reshape(4,B,HD)
  hidB[i] = (__bf16)lat;
  if (i < BB * DIN) inp[i] = (__bf16)0.0f;   // decoder step-0 input = zeros
}

extern "C" void kernel_launch(void* const* d_in, const int* in_sizes, int n_in,
                              void* d_out, int out_size, void* d_ws, size_t ws_size,
                              hipStream_t stream) {
  (void)in_sizes; (void)n_in; (void)out_size; (void)ws_size;
  const float* x     = (const float*)d_in[0];
  const float* h0    = (const float*)d_in[1];
  const float* eps   = (const float*)d_in[2];
  const float* eWih0 = (const float*)d_in[3];
  const float* eWhh0 = (const float*)d_in[4];
  const float* ebih0 = (const float*)d_in[5];
  const float* ebhh0 = (const float*)d_in[6];
  const float* eWih1 = (const float*)d_in[7];
  const float* eWhh1 = (const float*)d_in[8];
  const float* ebih1 = (const float*)d_in[9];
  const float* ebhh1 = (const float*)d_in[10];
  const float* dWih0 = (const float*)d_in[11];
  const float* dWhh0 = (const float*)d_in[12];
  const float* dbih0 = (const float*)d_in[13];
  const float* dbhh0 = (const float*)d_in[14];
  const float* dWih1 = (const float*)d_in[15];
  const float* dWhh1 = (const float*)d_in[16];
  const float* dbih1 = (const float*)d_in[17];
  const float* dbhh1 = (const float*)d_in[18];
  const float* linW  = (const float*)d_in[19];
  const float* linB  = (const float*)d_in[20];
  float* out = (float*)d_out;

  // --- workspace carve-out (256B aligned) ---
  char* ws = (char*)d_ws;
  size_t cur = 0;
  auto alloc = [&](size_t bytes) -> void* {
    cur = (cur + 255) & ~(size_t)255;
    void* p = ws + cur;
    cur += bytes;
    return p;
  };
  __bf16* wEih0 = (__bf16*)alloc((size_t)2 * 1536 * 128  * 2);
  __bf16* wEhh0 = (__bf16*)alloc((size_t)2 * 1536 * 512  * 2);
  __bf16* wEih1 = (__bf16*)alloc((size_t)2 * 1536 * 1024 * 2);
  __bf16* wEhh1 = (__bf16*)alloc((size_t)2 * 1536 * 512  * 2);
  __bf16* wDih0 = (__bf16*)alloc((size_t)2 * 768 * 128 * 2);
  __bf16* wDhh0 = (__bf16*)alloc((size_t)2 * 768 * 256 * 2);
  __bf16* wDih1 = (__bf16*)alloc((size_t)2 * 768 * 512 * 2);
  __bf16* wDhh1 = (__bf16*)alloc((size_t)2 * 768 * 256 * 2);
  __bf16* wLin  = (__bf16*)alloc((size_t)128 * 512 * 2);
  __bf16* xbf   = (__bf16*)alloc((size_t)BB * TT * DIN * 2);
  __bf16* h0bf  = (__bf16*)alloc((size_t)4 * BB * HHE * 2);
  __bf16* ys0b  = (__bf16*)alloc((size_t)TT * BB * 1024 * 2);    // layer0 output seq
  __bf16* hEb   = (__bf16*)alloc((size_t)4 * 2 * BB * HHE * 2);  // [lyr*2+par][2][B][512]
  float*  hEf   = (float*) alloc((size_t)4 * 2 * BB * HHE * 4);
  __bf16* hidb  = (__bf16*)alloc((size_t)2 * 4 * BB * HHD * 2);  // [par][4][B][256]
  float*  hidf  = (float*) alloc((size_t)2 * 4 * BB * HHD * 4);
  __bf16* x1b   = (__bf16*)alloc((size_t)BB * 512 * 2);
  __bf16* y1b   = (__bf16*)alloc((size_t)BB * 512 * 2);
  __bf16* inpb  = (__bf16*)alloc((size_t)BB * DIN * 2);

  auto cvt = [&](const float* s, __bf16* d, int n) {
    f32_to_bf16_k<<<dim3((n + 255) / 256), dim3(256), 0, stream>>>(s, d, n);
  };
  cvt(eWih0, wEih0, 2 * 1536 * 128);
  cvt(eWhh0, wEhh0, 2 * 1536 * 512);
  cvt(eWih1, wEih1, 2 * 1536 * 1024);
  cvt(eWhh1, wEhh1, 2 * 1536 * 512);
  cvt(dWih0, wDih0, 2 * 768 * 128);
  cvt(dWhh0, wDhh0, 2 * 768 * 256);
  cvt(dWih1, wDih1, 2 * 768 * 512);
  cvt(dWhh1, wDhh1, 2 * 768 * 256);
  cvt(linW,  wLin,  128 * 512);
  cvt(x,     xbf,   BB * TT * DIN);
  cvt(h0,    h0bf,  4 * BB * HHE);

  const dim3 blk(256);
  const dim3 gridE(8, 4, 2);   // B/32 x 512/128 x 2 dirs
  const dim3 gridD(8, 2, 2);   // B/32 x 256/128 x 2 dirs
  const size_t eBuf = (size_t)2 * BB * HHE;   // one [2][B][512] panel
  const size_t dBuf = (size_t)4 * BB * HHD;   // one [4][B][256] panel

  // ---- encoder layer 0 (bidirectional, writes ys0) ----
  const float*  hcF = h0;       // initial state straight from input [4,B,512]
  const __bf16* hcB = h0bf;
  for (int t = 0; t < TT; ++t) {
    float*  hnF = hEf + (size_t)(0 * 2 + (t & 1)) * eBuf;
    __bf16* hnB = hEb + (size_t)(0 * 2 + (t & 1)) * eBuf;
    gru_step_fused<<<gridE, blk, 0, stream>>>(
        xbf, (long)DIN, (long)TT * DIN, t, TT,
        wEih0, wEhh0, ebih0, ebhh0,
        hcF, hcB, hnF, hnB,
        ys0b, (long)BB * 1024, 1024L, DIN, HHE);
    hcF = hnF; hcB = hnB;
  }
  const float* hE0fin = hcF;

  // ---- encoder layer 1 (input = ys0, no ys output, only final h) ----
  hcF = h0 + (size_t)2 * BB * HHE;
  hcB = h0bf + (size_t)2 * BB * HHE;
  for (int t = 0; t < TT; ++t) {
    float*  hnF = hEf + (size_t)(1 * 2 + (t & 1)) * eBuf;
    __bf16* hnB = hEb + (size_t)(1 * 2 + (t & 1)) * eBuf;
    gru_step_fused<<<gridE, blk, 0, stream>>>(
        ys0b, (long)BB * 1024, 1024L, t, TT,
        wEih1, wEhh1, ebih1, ebhh1,
        hcF, hcB, hnF, hnB,
        (__bf16*)nullptr, 0L, 0L, 1024, HHE);
    hcF = hnF; hcB = hnB;
  }
  const float* hE1fin = hcF;

  // ---- latent: q outputs, reparam, decoder hidden init, zero input ----
  latent_k<<<dim3((BB * 1024 + 255) / 256), blk, 0, stream>>>(
      hE0fin, hE1fin, eps, out, hidf, hidb, inpb);

  // ---- decoder: per step, layer0 -> layer1 -> linear (feedback) ----
  float* rec = out + QN;
  for (int t = 0; t < TT; ++t) {
    const int p = t & 1;
    float*  cF = hidf + (size_t)p * dBuf;
    __bf16* cB = hidb + (size_t)p * dBuf;
    float*  nF = hidf + (size_t)(1 - p) * dBuf;
    __bf16* nB = hidb + (size_t)(1 - p) * dBuf;
    gru_step_fused<<<gridD, blk, 0, stream>>>(
        inpb, 0L, (long)DIN, 0, 1,
        wDih0, wDhh0, dbih0, dbhh0,
        cF, cB, nF, nB,
        x1b, 0L, 512L, DIN, HHD);
    gru_step_fused<<<gridD, blk, 0, stream>>>(
        x1b, 0L, 512L, 0, 1,
        wDih1, wDhh1, dbih1, dbhh1,
        cF + 2 * (size_t)BB * HHD, cB + 2 * (size_t)BB * HHD,
        nF + 2 * (size_t)BB * HHD, nB + 2 * (size_t)BB * HHD,
        y1b, 0L, 512L, 512, HHD);
    linear_step<<<dim3(16), blk, 0, stream>>>(y1b, wLin, linB, rec, inpb, t);
  }
}
